// SupervisedContrastiveLoss_29214367547598
// MI455X (gfx1250) — compile-verified, matrix operand order checked by reading, not run on hardware
//
#include <hip/hip_runtime.h>
#include <hip/hip_bf16.h>

typedef __attribute__((ext_vector_type(16))) _Float16 v16h;
typedef __attribute__((ext_vector_type(8)))  _Float16 v8h;
typedef __attribute__((ext_vector_type(4)))  _Float16 v4h;
typedef __attribute__((ext_vector_type(8)))  float    v8f;
typedef __attribute__((ext_vector_type(4)))  float    v4f;

#define INV_TEMP 10.0f
#define EPSN     1e-8f

// ---------------------------------------------------------------------------
// Kernel 1: L2-normalize one row (eps-clamped) and pack f32 -> f16.
// One block per row; 256 threads x 4 floats = 1024 = D.
// ---------------------------------------------------------------------------
__global__ void cl_normalize_f16(const float* __restrict__ z,
                                 _Float16* __restrict__ out, int D) {
    const int row  = blockIdx.x;
    const int tid  = threadIdx.x;
    const int lane = tid & 31;
    const int wid  = tid >> 5;

    const float* rp = z + (size_t)row * D;

    float ss = 0.0f;
    for (int j = tid * 4; j < D; j += blockDim.x * 4) {
        v4f v = *(const v4f*)(rp + j);
        ss += v.x * v.x + v.y * v.y + v.z * v.z + v.w * v.w;
    }
    // wave32 reduce
    ss += __shfl_xor(ss, 1);
    ss += __shfl_xor(ss, 2);
    ss += __shfl_xor(ss, 4);
    ss += __shfl_xor(ss, 8);
    ss += __shfl_xor(ss, 16);

    __shared__ float red[8];
    if (lane == 0) red[wid] = ss;
    __syncthreads();
    if (tid < 8) {
        float t = red[tid];
        t += __shfl_xor(t, 1);
        t += __shfl_xor(t, 2);
        t += __shfl_xor(t, 4);
        if (tid == 0) red[0] = t;
    }
    __syncthreads();
    const float scale = 1.0f / fmaxf(sqrtf(red[0]), EPSN);

    _Float16* op = out + (size_t)row * D;
    for (int j = tid * 4; j < D; j += blockDim.x * 4) {
        v4f v = *(const v4f*)(rp + j);
        v4h h;
        h.x = (_Float16)(v.x * scale);
        h.y = (_Float16)(v.y * scale);
        h.z = (_Float16)(v.z * scale);
        h.w = (_Float16)(v.w * scale);
        *(v4h*)(op + j) = h;  // 64-bit store
    }
}

// ---------------------------------------------------------------------------
// Kernel 2: diag[i] = <z1n[i], z2n[i]> * (1/T)   (keeps GEMM epilogue lean)
// ---------------------------------------------------------------------------
__global__ void cl_diag(const _Float16* __restrict__ z1h,
                        const _Float16* __restrict__ z2h,
                        float* __restrict__ diagv, int D) {
    const int row  = blockIdx.x;
    const int tid  = threadIdx.x;
    const int lane = tid & 31;
    const int wid  = tid >> 5;

    const _Float16* ap = z1h + (size_t)row * D;
    const _Float16* bp = z2h + (size_t)row * D;

    float acc = 0.0f;
    for (int j = tid * 4; j < D; j += blockDim.x * 4) {
        v4h a = *(const v4h*)(ap + j);
        v4h b = *(const v4h*)(bp + j);
        acc += (float)a.x * (float)b.x + (float)a.y * (float)b.y +
               (float)a.z * (float)b.z + (float)a.w * (float)b.w;
    }
    acc += __shfl_xor(acc, 1);
    acc += __shfl_xor(acc, 2);
    acc += __shfl_xor(acc, 4);
    acc += __shfl_xor(acc, 8);
    acc += __shfl_xor(acc, 16);

    __shared__ float red[8];
    if (lane == 0) red[wid] = acc;
    __syncthreads();
    if (tid == 0) {
        float t = 0.0f;
        for (int i = 0; i < 8; ++i) t += red[i];
        diagv[row] = t * INV_TEMP;
    }
}

// ---------------------------------------------------------------------------
// Fragment loader for v_wmma_f32_16x16x32_f16.
// 16-bit A-matrix 16x32 layout (CDNA5 ISA 7.12.2):
//   lanes 0-15  (row M = lane):     halves 0-7 = K 0..7,  halves 8-15 = K 16..23
//   lanes 16-31 (row M = lane-16):  halves 0-7 = K 8..15, halves 8-15 = K 24..31
// `p` must already point at row base + (hi ? 8 : 0) within the 32-wide K slab.
// Two global_load_b128 per fragment per lane.
// ---------------------------------------------------------------------------
__device__ __forceinline__ v16h load_frag(const _Float16* p) {
    union { v16h v; v8h h[2]; } u;
    u.h[0] = *(const v8h*)(p);        // K + {0..7}
    u.h[1] = *(const v8h*)(p + 16);   // K + 16 + {0..7}
    return u.v;
}

__device__ __forceinline__ v8f wmma16x16x32(v16h a, v16h b, v8f c) {
    return __builtin_amdgcn_wmma_f32_16x16x32_f16(
        /*neg_a=*/false, a, /*neg_b=*/false, b,
        /*c_mod=*/(short)0, c, /*reuse_a=*/false, /*reuse_b=*/false);
}

// ---------------------------------------------------------------------------
// Kernel 3: fused GEMM (z1n @ z2n^T / T) + online exp-row-sums.
// Block = 256 threads = 8 waves arranged 2(M) x 4(N); block tile 128 x 256.
// Each wave computes a 64x64 output patch as 4x4 WMMA 16x16 tiles:
// 16 WMMAs per 16 b128 loads per K-step -> 32 FLOP/B from L2.
// logits never touch memory.
// ---------------------------------------------------------------------------
__global__ void __launch_bounds__(256)
cl_gemm_lse(const _Float16* __restrict__ z1h, const _Float16* __restrict__ z2h,
            float* __restrict__ rowsum, int N, int D) {
    const int lane = threadIdx.x & 31;
    const int w    = threadIdx.x >> 5;
    const int r    = lane & 15;      // row/col-within-tile index
    const int hi   = lane >> 4;      // half-wave select (K-slab swizzle)

    const int waveM = w & 1;         // 0..1
    const int waveN = w >> 1;        // 0..3
    const int mBase = blockIdx.y * 128 + waveM * 64;
    const int nBase = blockIdx.x * 256 + waveN * 64;

    // Per-lane fragment base pointers (A rows of z1n; "B columns" = rows of
    // z2n with the same swizzle, since logits = A * B^T).
    const _Float16* ap[4];
    const _Float16* bp[4];
#pragma unroll
    for (int t = 0; t < 4; ++t) {
        ap[t] = z1h + (size_t)(mBase + 16 * t + r) * D + hi * 8;
        bp[t] = z2h + (size_t)(nBase + 16 * t + r) * D + hi * 8;
    }

    v8f acc[4][4];
#pragma unroll
    for (int i = 0; i < 4; ++i)
#pragma unroll
        for (int j = 0; j < 4; ++j) acc[i][j] = (v8f){};

    for (int k = 0; k < D; k += 32) {
        v16h a[4], b[4];
#pragma unroll
        for (int t = 0; t < 4; ++t) a[t] = load_frag(ap[t] + k);
#pragma unroll
        for (int t = 0; t < 4; ++t) b[t] = load_frag(bp[t] + k);
#pragma unroll
        for (int i = 0; i < 4; ++i)
#pragma unroll
            for (int j = 0; j < 4; ++j)
                acc[i][j] = wmma16x16x32(a[i], b[j], acc[i][j]);
    }

    // Epilogue: exp row-sums.  C/D layout: VGPR i, lanes 0-15 -> (M=i, N=lane),
    // lanes 16-31 -> (M=i+8, N=lane-16).
#pragma unroll
    for (int tm = 0; tm < 4; ++tm) {
#pragma unroll
        for (int i = 0; i < 8; ++i) {
            const int row = mBase + tm * 16 + i + hi * 8;
            float s = 0.0f;
#pragma unroll
            for (int tn = 0; tn < 4; ++tn)
                s += __expf(acc[tm][tn][i] * INV_TEMP);
            // reduce across the 16-lane column group (rows identical there)
            s += __shfl_xor(s, 1);
            s += __shfl_xor(s, 2);
            s += __shfl_xor(s, 4);
            s += __shfl_xor(s, 8);
            if (r == 0) atomicAdd(&rowsum[row], s);
        }
    }
}

// ---------------------------------------------------------------------------
// Kernel 4: loss = -mean(diag_i - log(rowsum_i))
// ---------------------------------------------------------------------------
__global__ void cl_finalize(const float* __restrict__ rowsum,
                            const float* __restrict__ diagv,
                            float* __restrict__ out, int N) {
    const int tid  = threadIdx.x;
    const int lane = tid & 31;
    const int wid  = tid >> 5;

    float acc = 0.0f;
    for (int i = tid; i < N; i += blockDim.x)
        acc += diagv[i] - logf(rowsum[i]);

    acc += __shfl_xor(acc, 1);
    acc += __shfl_xor(acc, 2);
    acc += __shfl_xor(acc, 4);
    acc += __shfl_xor(acc, 8);
    acc += __shfl_xor(acc, 16);

    __shared__ float red[8];
    if (lane == 0) red[wid] = acc;
    __syncthreads();
    if (tid == 0) {
        float t = 0.0f;
        for (int i = 0; i < 8; ++i) t += red[i];
        out[0] = -t / (float)N;
    }
}

// ---------------------------------------------------------------------------
extern "C" void kernel_launch(void* const* d_in, const int* in_sizes, int n_in,
                              void* d_out, int out_size, void* d_ws, size_t ws_size,
                              hipStream_t stream) {
    const float* z1 = (const float*)d_in[0];
    const float* z2 = (const float*)d_in[1];
    const int D = 1024;
    const int N = in_sizes[0] / D;   // 8192

    // Workspace: z1h (N*D f16) | z2h (N*D f16) | rowsum (N f32) | diag (N f32)
    _Float16* z1h = (_Float16*)d_ws;
    _Float16* z2h = z1h + (size_t)N * D;
    float* rowsum = (float*)(z2h + (size_t)N * D);
    float* diagv  = rowsum + N;

    hipMemsetAsync(rowsum, 0, (size_t)N * sizeof(float), stream);

    cl_normalize_f16<<<dim3(N), 256, 0, stream>>>(z1, z1h, D);
    cl_normalize_f16<<<dim3(N), 256, 0, stream>>>(z2, z2h, D);
    cl_diag<<<dim3(N), 256, 0, stream>>>(z1h, z2h, diagv, D);

    dim3 grid(N / 256, N / 128);     // block tile: 128 (M) x 256 (N)
    cl_gemm_lse<<<grid, 256, 0, stream>>>(z1h, z2h, rowsum, N, D);

    cl_finalize<<<1, 256, 0, stream>>>(rowsum, diagv, (float*)d_out, N);
}
